// MultiHeadAttention_7249904796457
// MI455X (gfx1250) — matrix operand order repaired and checked by reading, not compiled, not run on hardware
//
#include <hip/hip_runtime.h>
#include <hip/hip_bf16.h>

typedef __attribute__((ext_vector_type(16))) __bf16 v16bf;
typedef __attribute__((ext_vector_type(8)))  __bf16 v8bf;
typedef __attribute__((ext_vector_type(8)))  float  v8f;
typedef __attribute__((ext_vector_type(4)))  unsigned int u32x4;
typedef __attribute__((ext_vector_type(8)))  int i32x8;
typedef __attribute__((ext_vector_type(4)))  int i32x4;

#define B_  4
#define S_  2048
#define D_  1024
#define H_  16
#define HD_ 64

#if defined(__gfx1250__) && __has_builtin(__builtin_amdgcn_tensor_load_to_lds) && \
    __has_builtin(__builtin_amdgcn_s_wait_tensorcnt)
#define USE_TDM 1
#else
#define USE_TDM 0
#endif

static __device__ inline v16bf cat8(v8bf lo, v8bf hi) {
  return __builtin_shufflevector(lo, hi, 0,1,2,3,4,5,6,7,8,9,10,11,12,13,14,15);
}

static __device__ inline v8f wmma_bf16(v16bf a, v16bf b, v8f c) {
  return __builtin_amdgcn_wmma_f32_16x16x32_bf16(false, a, false, b, (short)0, c, false, false);
}

static __device__ inline float half16_max(float v) {
  v = fmaxf(v, __shfl_xor(v, 1, 32));
  v = fmaxf(v, __shfl_xor(v, 2, 32));
  v = fmaxf(v, __shfl_xor(v, 4, 32));
  v = fmaxf(v, __shfl_xor(v, 8, 32));
  return v;
}
static __device__ inline float half16_sum(float v) {
  v += __shfl_xor(v, 1, 32);
  v += __shfl_xor(v, 2, 32);
  v += __shfl_xor(v, 4, 32);
  v += __shfl_xor(v, 8, 32);
  return v;
}

#if USE_TDM
// LDS byte offset of a shared-memory pointer (generic -> addrspace(3)).
#define LDS_OFF(p) ((unsigned)(unsigned long long)(__attribute__((address_space(3))) const void*)(p))

// Issue one TDM 2D tile load: global (strideElems row pitch, bf16) -> LDS (packed).
// Descriptor packed per CDNA5 ISA ch.8 (D# groups 0/1); groups 2/3 unused (2D).
static __device__ inline void tdm_load_2d(unsigned ldsOff, const void* gptr,
                                          int tileW, int tileH, int strideElems) {
  unsigned long long ga = (unsigned long long)gptr;
  u32x4 g0;
  g0.x = 1u;                                       // count=1, user descriptor
  g0.y = ldsOff;                                   // lds_addr[63:32]
  g0.z = (unsigned)(ga & 0xffffffffu);             // global_addr[95:64]
  g0.w = (unsigned)((ga >> 32) & 0x01ffffffu) | (2u << 30);  // addr hi + type=2
  i32x8 g1;
  g1[0] = (1 << 16);                               // data_size=1 (2 bytes), no multicast
  g1[1] = (tileW & 0xffff) << 16;                  // tensor_dim0 low16 @ [63:48]
  g1[2] = ((tileW >> 16) & 0xffff) | ((tileH & 0xffff) << 16);  // t0 hi, t1 lo
  g1[3] = ((tileH >> 16) & 0xffff) | (tileW << 16);             // t1 hi, tile_dim0
  g1[4] = (tileH & 0xffff);                        // tile_dim1 (tile_dim2=0)
  g1[5] = strideElems;                             // tensor_dim0_stride low32
  g1[6] = 0;
  g1[7] = 0;
  i32x4 gz = {0, 0, 0, 0};
#if __clang_major__ >= 23
  i32x8 gz8 = {0, 0, 0, 0, 0, 0, 0, 0};
  __builtin_amdgcn_tensor_load_to_lds(g0, g1, gz, gz, gz8, 0);
#else
  __builtin_amdgcn_tensor_load_to_lds(g0, g1, gz, gz, 0);
#endif
}
#endif  // USE_TDM

// ---------------------------------------------------------------------------
// Conversion kernels
// ---------------------------------------------------------------------------
__global__ void f32_to_bf16_kernel(const float* __restrict__ in,
                                   __bf16* __restrict__ out, int n) {
  int i = blockIdx.x * blockDim.x + threadIdx.x;
  if (i < n) out[i] = (__bf16)in[i];
}

__global__ void transpose_f32_to_bf16_kernel(const float* __restrict__ in,
                                             __bf16* __restrict__ out,
                                             int R, int C) {
  int i = blockIdx.x * blockDim.x + threadIdx.x;
  if (i < R * C) {
    int r = i / C, c = i % C;
    out[(size_t)c * R + r] = (__bf16)in[i];
  }
}

// ---------------------------------------------------------------------------
// bf16 WMMA GEMM: C[M,N] = A[M,K] @ BT[N,K]^T
// Block 256 (8 waves): block tile 128x128, wave tile 32(M)x64(N), K-stage 64.
// OUT_MODE 0: bf16 row-major   1: V-transposed [B][H][HD][S]   2: f32 + bias
// ---------------------------------------------------------------------------
static __device__ inline void gemm_compute_stage(const __bf16* __restrict__ As,
                                                 const __bf16* __restrict__ Bs,
                                                 int wm, int wn, int lh, int lm,
                                                 v8f acc[2][4]) {
#pragma unroll
  for (int c = 0; c < 2; ++c) {
    v16bf af[2];
#pragma unroll
    for (int t = 0; t < 2; ++t) {
      int row = wm * 32 + t * 16 + lm;
      af[t] = *(const v16bf*)&As[row * 64 + c * 32 + lh * 16];
    }
#pragma unroll
    for (int u = 0; u < 4; ++u) {
      int col = wn * 64 + u * 16 + lm;
      v16bf bfr = *(const v16bf*)&Bs[col * 64 + c * 32 + lh * 16];
#pragma unroll
      for (int t = 0; t < 2; ++t) acc[t][u] = wmma_bf16(af[t], bfr, acc[t][u]);
    }
  }
}

template <int OUT_MODE>
__global__ __launch_bounds__(256)
void gemm_bf16_kernel(const __bf16* __restrict__ A, const __bf16* __restrict__ BT,
                      void* __restrict__ Cout, const float* __restrict__ bias,
                      int M, int N, int K) {
  __shared__ __align__(32) __bf16 As[2][128 * 64];   // 2 x 16KB
  __shared__ __align__(32) __bf16 Bs[2][128 * 64];   // 2 x 16KB

  const int tid  = threadIdx.x;
  const int wave = tid >> 5;
  const int lane = tid & 31;
  const int lh   = lane >> 4;
  const int lm   = lane & 15;
  const int wm   = wave >> 1;   // 0..3 (M)
  const int wn   = wave & 1;    // 0..1 (N)

  const int mBase = blockIdx.y * 128;
  const int nBase = blockIdx.x * 128;

  v8f acc[2][4] = {{{}, {}, {}, {}}, {{}, {}, {}, {}}};
  const int nStages = K / 64;

#if USE_TDM
  if (wave == 0) {
    tdm_load_2d(LDS_OFF(&As[0][0]), &A[(size_t)mBase * K], 64, 128, K);
    tdm_load_2d(LDS_OFF(&Bs[0][0]), &BT[(size_t)nBase * K], 64, 128, K);
  }
  for (int s = 0; s < nStages; ++s) {
    const int cur = s & 1;
    if (wave == 0) {
      if (s + 1 < nStages) {
        int k0 = (s + 1) * 64;
        tdm_load_2d(LDS_OFF(&As[cur ^ 1][0]), &A[(size_t)mBase * K + k0], 64, 128, K);
        tdm_load_2d(LDS_OFF(&Bs[cur ^ 1][0]), &BT[(size_t)nBase * K + k0], 64, 128, K);
        __builtin_amdgcn_s_wait_tensorcnt(2);  // in-order: stage s's two loads done
      } else {
        __builtin_amdgcn_s_wait_tensorcnt(0);
      }
    }
    __syncthreads();
    gemm_compute_stage(&As[cur][0], &Bs[cur][0], wm, wn, lh, lm, acc);
    __syncthreads();
  }
#else
  for (int s = 0; s < nStages; ++s) {
    const int k0 = s * 64;
#pragma unroll
    for (int c = tid; c < 1024; c += 256) {  // A: 128 rows x 64 k, v8bf chunks
      int row = c >> 3, kc = (c & 7) * 8;
      *(v8bf*)&As[0][row * 64 + kc] = *(const v8bf*)&A[(size_t)(mBase + row) * K + k0 + kc];
    }
#pragma unroll
    for (int c = tid; c < 1024; c += 256) {  // B
      int row = c >> 3, kc = (c & 7) * 8;
      *(v8bf*)&Bs[0][row * 64 + kc] = *(const v8bf*)&BT[(size_t)(nBase + row) * K + k0 + kc];
    }
    __syncthreads();
    gemm_compute_stage(&As[0][0], &Bs[0][0], wm, wn, lh, lm, acc);
    __syncthreads();
  }
#endif

  // epilogue: 16x16 C tile -> VGPR j holds (row = j + 8*lh, col = lm)
#pragma unroll
  for (int t = 0; t < 2; ++t) {
#pragma unroll
    for (int u = 0; u < 4; ++u) {
#pragma unroll
      for (int j = 0; j < 8; ++j) {
        int grow = mBase + wm * 32 + t * 16 + lh * 8 + j;
        int gcol = nBase + wn * 64 + u * 16 + lm;
        float v  = acc[t][u][j];
        if (OUT_MODE == 0) {
          ((__bf16*)Cout)[(size_t)grow * N + gcol] = (__bf16)v;
        } else if (OUT_MODE == 1) {
          int bidx = grow >> 11, s = grow & 2047;
          int h = gcol >> 6, d = gcol & 63;
          ((__bf16*)Cout)[(((size_t)(bidx * H_ + h)) * HD_ + d) * S_ + s] = (__bf16)v;
        } else {
          ((float*)Cout)[(size_t)grow * N + gcol] = v + bias[gcol];
        }
      }
    }
  }
}

// ---------------------------------------------------------------------------
// Flash attention. Block 256 (8 waves) handles 128 query rows of one (b,h);
// K/V 64-key tiles staged in LDS and shared by all 8 waves.
// Q,K: bf16 [B*S, D] (col = h*HD+d).  Vt: bf16 [B][H][HD][S].  ctx: bf16 [B*S, D].
// ---------------------------------------------------------------------------
__global__ __launch_bounds__(256)
void flash_attn_kernel(const __bf16* __restrict__ Q, const __bf16* __restrict__ Kb,
                       const __bf16* __restrict__ Vt, __bf16* __restrict__ ctx) {
  __shared__ __align__(32) __bf16 Ks[64 * 64];        // [key][d]   8KB
  __shared__ __align__(32) __bf16 Vs[64 * 64];        // [d][key]   8KB
  __shared__ __align__(32) __bf16 pShm[8][16][64];    // per-wave P 16KB

  const int tid  = threadIdx.x;
  const int wave = tid >> 5;
  const int lane = tid & 31;
  const int lh   = lane >> 4;
  const int lm   = lane & 15;

  const int b = blockIdx.z;
  const int h = blockIdx.y;
  const int qBase = blockIdx.x * 128 + wave * 16;

  // Q A-fragments (hd=64 -> two 16x32 k-chunks)
  const __bf16* qrow = Q + ((size_t)(b * S_ + qBase + lm)) * D_ + h * HD_;
  v16bf qa[2];
#pragma unroll
  for (int c = 0; c < 2; ++c)
    qa[c] = *(const v16bf*)(qrow + c * 32 + lh * 16);

  float m[8], l[8];
#pragma unroll
  for (int j = 0; j < 8; ++j) { m[j] = -INFINITY; l[j] = 0.f; }
  v8f accO[4] = {{}, {}, {}, {}};

  const __bf16* kbase = Kb + ((size_t)b * S_) * D_ + h * HD_;     // +key*D_
  const __bf16* vbase = Vt + ((size_t)(b * H_ + h)) * HD_ * S_;   // +d*S_+key
  const float scale = 0.125f;  // 1/sqrt(64)

  for (int kt = 0; kt < S_; kt += 64) {
#if USE_TDM
    if (wave == 0) {
      tdm_load_2d(LDS_OFF(&Ks[0]), kbase + (size_t)kt * D_, 64, 64, D_);
      tdm_load_2d(LDS_OFF(&Vs[0]), vbase + kt, 64, 64, S_);
      __builtin_amdgcn_s_wait_tensorcnt(0);
    }
#else
#pragma unroll
    for (int c = tid; c < 512; c += 256) {
      int row = c >> 3, kc = (c & 7) * 8;
      *(v8bf*)&Ks[row * 64 + kc] = *(const v8bf*)&kbase[(size_t)(kt + row) * D_ + kc];
    }
#pragma unroll
    for (int c = tid; c < 512; c += 256) {
      int row = c >> 3, kc = (c & 7) * 8;
      *(v8bf*)&Vs[row * 64 + kc] = *(const v8bf*)&vbase[(size_t)row * S_ + kt + kc];
    }
#endif
    __syncthreads();

    // scores: 16 q-rows x 64 keys = 4 x (16x16), K-dim 64 = 2 wmma each
    v8f sc[4] = {{}, {}, {}, {}};
#pragma unroll
    for (int g = 0; g < 4; ++g)
#pragma unroll
      for (int c = 0; c < 2; ++c) {
        v16bf kf = *(const v16bf*)&Ks[(g * 16 + lm) * 64 + c * 32 + lh * 16];
        sc[g] = wmma_bf16(qa[c], kf, sc[g]);
      }

    // online softmax (row = j + 8*lh, stats replicated over 16-lane half)
    float fj[8];
#pragma unroll
    for (int j = 0; j < 8; ++j) {
      float s0 = sc[0][j] * scale, s1 = sc[1][j] * scale;
      float s2 = sc[2][j] * scale, s3 = sc[3][j] * scale;
      float mt = half16_max(fmaxf(fmaxf(s0, s1), fmaxf(s2, s3)));
      float mn = fmaxf(m[j], mt);
      float f  = __expf(m[j] - mn);
      float p0 = __expf(s0 - mn), p1 = __expf(s1 - mn);
      float p2 = __expf(s2 - mn), p3 = __expf(s3 - mn);
      float r  = half16_sum(p0 + p1 + p2 + p3);
      l[j] = l[j] * f + r;
      m[j] = mn;
      fj[j] = f;
      int row = lh * 8 + j;
      pShm[wave][row][lm]      = (__bf16)p0;
      pShm[wave][row][16 + lm] = (__bf16)p1;
      pShm[wave][row][32 + lm] = (__bf16)p2;
      pShm[wave][row][48 + lm] = (__bf16)p3;
    }
#pragma unroll
    for (int t = 0; t < 4; ++t)
#pragma unroll
      for (int j = 0; j < 8; ++j) accO[t][j] *= fj[j];

    __syncthreads();

    // O += P(16x64) @ V(64x64): reload P as A-fragments, V from LDS
    v16bf pa[2];
#pragma unroll
    for (int c = 0; c < 2; ++c)
      pa[c] = *(const v16bf*)&pShm[wave][lm][c * 32 + lh * 16];
#pragma unroll
    for (int t = 0; t < 4; ++t)
#pragma unroll
      for (int c = 0; c < 2; ++c) {
        v16bf vf = *(const v16bf*)&Vs[(t * 16 + lm) * 64 + c * 32 + lh * 16];
        accO[t] = wmma_bf16(pa[c], vf, accO[t]);
      }
    __syncthreads();
  }

  // normalize and store ctx
#pragma unroll
  for (int t = 0; t < 4; ++t)
#pragma unroll
    for (int j = 0; j < 8; ++j) {
      int row = lh * 8 + j;
      float v = accO[t][j] / l[j];
      ctx[((size_t)(b * S_ + qBase + row)) * D_ + h * HD_ + t * 16 + lm] = (__bf16)v;
    }
}

// ---------------------------------------------------------------------------
extern "C" void kernel_launch(void* const* d_in, const int* in_sizes, int n_in,
                              void* d_out, int out_size, void* d_ws, size_t ws_size,
                              hipStream_t stream) {
  const float* x  = (const float*)d_in[0];
  const float* Wq = (const float*)d_in[1];
  const float* Wk = (const float*)d_in[2];
  const float* Wv = (const float*)d_in[3];
  const float* Wo = (const float*)d_in[4];
  const float* bo = (const float*)d_in[5];

  const int M = B_ * S_;  // 8192
  const int N = D_;       // 1024
  const int K = D_;       // 1024

  char* w = (char*)d_ws;
  const size_t MB = 1024ull * 1024ull;
  __bf16* xb  = (__bf16*)(w + 0);
  __bf16* WqT = (__bf16*)(w + 16 * MB);
  __bf16* WkT = (__bf16*)(w + 18 * MB);
  __bf16* WvT = (__bf16*)(w + 20 * MB);
  __bf16* WoT = (__bf16*)(w + 22 * MB);
  __bf16* Qb  = (__bf16*)(w + 24 * MB);
  __bf16* Kb  = (__bf16*)(w + 40 * MB);
  __bf16* Vt  = (__bf16*)(w + 56 * MB);  // [B][H][HD][S]
  __bf16* ctx = (__bf16*)(w + 72 * MB);

  {
    int n = M * K;
    f32_to_bf16_kernel<<<(n + 255) / 256, 256, 0, stream>>>(x, xb, n);
    int nw = K * N;
    int g = (nw + 255) / 256;
    transpose_f32_to_bf16_kernel<<<g, 256, 0, stream>>>(Wq, WqT, K, N);
    transpose_f32_to_bf16_kernel<<<g, 256, 0, stream>>>(Wk, WkT, K, N);
    transpose_f32_to_bf16_kernel<<<g, 256, 0, stream>>>(Wv, WvT, K, N);
    transpose_f32_to_bf16_kernel<<<g, 256, 0, stream>>>(Wo, WoT, K, N);
  }

  dim3 gemmGrid(N / 128, M / 128);  // (8, 64)
  gemm_bf16_kernel<0><<<gemmGrid, 256, 0, stream>>>(xb, WqT, (void*)Qb, nullptr, M, N, K);
  gemm_bf16_kernel<0><<<gemmGrid, 256, 0, stream>>>(xb, WkT, (void*)Kb, nullptr, M, N, K);
  gemm_bf16_kernel<1><<<gemmGrid, 256, 0, stream>>>(xb, WvT, (void*)Vt, nullptr, M, N, K);

  dim3 attnGrid(S_ / 128, H_, B_);  // (16, 16, 4)
  flash_attn_kernel<<<attnGrid, 256, 0, stream>>>(Qb, Kb, Vt, ctx);

  gemm_bf16_kernel<2><<<gemmGrid, 256, 0, stream>>>(ctx, WoT, d_out, bo, M, N, K);
}
